// Attention_10754598110072
// MI455X (gfx1250) — compile-verified, matrix operand order checked by reading, not run on hardware
//
#include <hip/hip_runtime.h>
#include <math.h>

#define Hd 512
#define Sd 2048
#define Bd 64

typedef __attribute__((ext_vector_type(16))) __bf16 v16bf;
typedef __attribute__((ext_vector_type(8)))  float  v8f;

// packed fp32x2 -> bf16x2 hardware convert (RNE), gfx1250
__device__ __forceinline__ unsigned cvt_pk_bf16(float lo, float hi) {
    unsigned r;
    asm("v_cvt_pk_bf16_f32 %0, %1, %2" : "=v"(r) : "v"(lo), "v"(hi));
    return r;
}

// CDNA5 async global->LDS DMA (ASYNCcnt tracked), 16B per lane
__device__ __forceinline__ void async_load_b128(unsigned lds_off, const void* gptr) {
    asm volatile("global_load_async_to_lds_b128 %0, %1, off"
                 :: "v"(lds_off), "v"((unsigned long long)(size_t)gptr)
                 : "memory");
}

// hardware tanh (gfx1250 trans op); v_nop covers the TRANS RAW hazard
__device__ __forceinline__ float htanh(float x) {
    float y;
    asm volatile("v_tanh_f32 %0, %1\n\tv_nop" : "=v"(y) : "v"(x));
    return y;
}

// ---------------------------------------------------------------------------
// Kernel 0: one-time W_w fp32 -> bf16 (re-used by all 4096 energy workgroups)
// ---------------------------------------------------------------------------
__global__ __launch_bounds__(256)
void convert_w_kernel(const float* __restrict__ Ww, unsigned short* __restrict__ Wbf)
{
    const int i = blockIdx.x * 256 + threadIdx.x;        // 65536 float4s
    float4 f = ((const float4*)Ww)[i];
    uint2 p;
    p.x = cvt_pk_bf16(f.x, f.y);
    p.y = cvt_pk_bf16(f.z, f.w);
    ((uint2*)Wbf)[i] = p;
}

// ---------------------------------------------------------------------------
// Kernel 1: u[b,k] = sum_h hidden[b,h] * U_w[k,h] + U_b[k]      (64x512)
// ---------------------------------------------------------------------------
__global__ __launch_bounds__(512)
void compute_u_kernel(const float* __restrict__ hidden,
                      const float* __restrict__ Uw,
                      const float* __restrict__ Ub,
                      float* __restrict__ u)
{
    __shared__ float sh[Hd];
    const int b = blockIdx.x;
    const int k = threadIdx.x;
    sh[k] = hidden[b * Hd + k];
    __syncthreads();
    const float4* wp = (const float4*)(Uw + (long)k * Hd);
    float acc = 0.f;
#pragma unroll 8
    for (int i = 0; i < Hd / 4; ++i) {
        float4 w4 = wp[i];
        acc += w4.x * sh[4 * i + 0] + w4.y * sh[4 * i + 1]
             + w4.z * sh[4 * i + 2] + w4.w * sh[4 * i + 3];
    }
    u[b * Hd + k] = acc + Ub[k];
}

// ---------------------------------------------------------------------------
// Kernel 2: fused energy[b,s] = V . tanh(u[b] + W_b + enc[b,s] @ W_w^T) + V_b
// 8 waves / workgroup, tile (b, 32 s-rows) x 512 cols, K-chunks of 32.
// B slab double-buffered via global_load_async_to_lds_b128; bf16 WMMA.
// Dynamic LDS: Bdb[2][512][32]bf16 (64K) | Abf[32][32]bf16 (2K) | sPart (1K)
// ---------------------------------------------------------------------------
#define SMEM_B_HALF (512 * 32 * 2)                 // 32 KB per buffer
#define SMEM_A_OFF  (2 * SMEM_B_HALF)              // 65536
#define SMEM_P_OFF  (SMEM_A_OFF + 32 * 32 * 2)     // 67584
#define SMEM_TOTAL  (SMEM_P_OFF + 8 * 32 * 4)      // 68608

__global__ __launch_bounds__(256)
void attn_energy_kernel(const float* __restrict__ enc,            // [B,S,H] fp32
                        const unsigned short* __restrict__ Wbf,   // [H,H] bf16
                        const float* __restrict__ Wb,             // [H]
                        const float* __restrict__ u,              // [B,H]
                        const float* __restrict__ Vw,             // [H]
                        const float* __restrict__ Vb,             // [1]
                        float* __restrict__ energy)               // [B,S]
{
    extern __shared__ __align__(128) char smem[];
    unsigned short* Bbuf = (unsigned short*)smem;                      // [2][512][32]
    unsigned short (*Abf)[32] = (unsigned short (*)[32])(smem + SMEM_A_OFF);
    float (*sPart)[32] = (float (*)[32])(smem + SMEM_P_OFF);
    const unsigned ldsBase = (unsigned)(size_t)smem;                   // LDS byte offset

    const int tid  = threadIdx.x;
    const int lane = tid & 31;
    const int wave = tid >> 5;
    const int half = lane >> 4;
    const int l15  = lane & 15;
    const int b    = blockIdx.y;
    const int s0   = blockIdx.x * 32;

    const int mt     = wave & 1;         // 16-row M-tile (0/1)
    const int ntBase = (wave >> 1) * 8;  // first of this wave's 8 N-tiles

    v8f acc[8] = {};
    const long encBase = ((long)b * Sd + s0) * Hd;

    // issue async copy of B chunk kc into buffer `buf`: 512 rows x 32 bf16
    // 2048 16-byte segments, 8 per thread
    auto issueB = [&](int kc, int buf) {
#pragma unroll
        for (int i = 0; i < 8; ++i) {
            int idx = tid + 256 * i;
            int n = idx >> 2, q = idx & 3;
            const unsigned short* g = Wbf + (long)n * Hd + kc * 32 + q * 8;
            unsigned lds = ldsBase + (unsigned)(buf * SMEM_B_HALF + (n * 32 + q * 8) * 2);
            async_load_b128(lds, g);
        }
    };

    issueB(0, 0);                                    // prologue

    for (int kc = 0; kc < Hd / 32; ++kc) {
        const int buf = kc & 1;

        // --- stage A: 32 rows x 32 k of enc, fp32 -> bf16 (packed hw convert)
        {
            int r = tid >> 3, q = tid & 7;           // 256 float4s
            float4 f = *(const float4*)(enc + encBase + (long)r * Hd + kc * 32 + 4 * q);
            uint2 p;
            p.x = cvt_pk_bf16(f.x, f.y);
            p.y = cvt_pk_bf16(f.z, f.w);
            *(uint2*)&Abf[r][4 * q] = p;
        }

        // --- prefetch next B chunk, then wait for the current one (8 in flight)
        if (kc < Hd / 32 - 1) {
            issueB(kc + 1, buf ^ 1);
            asm volatile("s_wait_asynccnt 0x8" ::: "memory");
        } else {
            asm volatile("s_wait_asynccnt 0x0" ::: "memory");
        }
        __syncthreads();

        // --- fragments per 16-bit wave32 layout:
        // frag dwords 0..3 = 16B at short-offset half*8; dwords 4..7 at 16+half*8
        union Frag { uint4 q[2]; v16bf v; };
        Frag af;
        {
            const unsigned short* ap = &Abf[mt * 16 + l15][0];
            af.q[0] = *(const uint4*)(ap + half * 8);
            af.q[1] = *(const uint4*)(ap + 16 + half * 8);
        }
        const unsigned short* Bb = Bbuf + buf * (512 * 32);
        Frag bfv[8];
#pragma unroll
        for (int t = 0; t < 8; ++t) {                // all fragments first:
            const unsigned short* bp = Bb + ((ntBase + t) * 16 + l15) * 32;
            bfv[t].q[0] = *(const uint4*)(bp + half * 8);
            bfv[t].q[1] = *(const uint4*)(bp + 16 + half * 8);
        }
#pragma unroll
        for (int t = 0; t < 8; ++t)                  // then WMMAs back-to-back
            acc[t] = __builtin_amdgcn_wmma_f32_16x16x32_bf16(
                false, af.v, false, bfv[t].v, (short)0, acc[t], false, false);
        __syncthreads();
    }

    // --- epilogue: e = tanh(acc + u[b,n] + W_b[n]) * V_w[n]; reduce over n
    // C layout: VGPR j, lane l -> row = j + 8*half (in M-tile), col = l&15
    // Accumulate over tiles t in-lane first, then one lane-reduction per j.
    float rsum[8];
#pragma unroll
    for (int j = 0; j < 8; ++j) rsum[j] = 0.f;

#pragma unroll
    for (int t = 0; t < 8; ++t) {
        const int ncol = (ntBase + t) * 16 + l15;
        const float un = u[b * Hd + ncol] + Wb[ncol];
        const float vn = Vw[ncol];
#pragma unroll
        for (int j = 0; j < 8; ++j)
            rsum[j] += htanh(acc[t][j] + un) * vn;
    }
#pragma unroll
    for (int j = 0; j < 8; ++j)
#pragma unroll
        for (int m = 1; m < 16; m <<= 1)             // sum 16 cols (stays in half)
            rsum[j] += __shfl_xor(rsum[j], m, 32);

    if (l15 == 0) {
#pragma unroll
        for (int j = 0; j < 8; ++j)
            sPart[wave][mt * 16 + (half << 3) + j] = rsum[j];
    }
    __syncthreads();

    if (tid < 32) {                                  // deterministic cross-wave sum
        const int r = tid, mtr = r >> 4;
        float e = Vb[0];
#pragma unroll
        for (int q = 0; q < 4; ++q)
            e += sPart[mtr + 2 * q][r];
        energy[(long)b * Sd + s0 + r] = e;
    }
}

// ---------------------------------------------------------------------------
// Kernel 3: softmax over S per batch row
// ---------------------------------------------------------------------------
__global__ __launch_bounds__(256)
void softmax_kernel(const float* __restrict__ energy, float* __restrict__ attn)
{
    __shared__ float red[256];
    const int b = blockIdx.x, tid = threadIdx.x;
    const float* e = energy + (long)b * Sd;

    float mx = -INFINITY;
#pragma unroll
    for (int i = 0; i < Sd / 256; ++i) mx = fmaxf(mx, e[tid + 256 * i]);
    red[tid] = mx; __syncthreads();
    for (int s = 128; s > 0; s >>= 1) {
        if (tid < s) red[tid] = fmaxf(red[tid], red[tid + s]);
        __syncthreads();
    }
    mx = red[0]; __syncthreads();

    float vals[Sd / 256];
    float sum = 0.f;
#pragma unroll
    for (int i = 0; i < Sd / 256; ++i) {
        float v = expf(e[tid + 256 * i] - mx);
        vals[i] = v; sum += v;
    }
    red[tid] = sum; __syncthreads();
    for (int s = 128; s > 0; s >>= 1) {
        if (tid < s) red[tid] += red[tid + s];
        __syncthreads();
    }
    const float inv = 1.f / red[0];
#pragma unroll
    for (int i = 0; i < Sd / 256; ++i)
        attn[(long)b * Sd + tid + 256 * i] = vals[i] * inv;
}

// ---------------------------------------------------------------------------
// Kernel 4: context partials over s-segments (memory-bound stream of enc)
// ---------------------------------------------------------------------------
__global__ __launch_bounds__(256)
void context_partial_kernel(const float* __restrict__ enc,
                            const float* __restrict__ attn,
                            float* __restrict__ part)   // [B,8,H]
{
    __shared__ float sA[256];
    const int b = blockIdx.y, seg = blockIdx.x, tid = threadIdx.x;
    sA[tid] = attn[(long)b * Sd + seg * 256 + tid];
    __syncthreads();
    const float* ep = enc + ((long)b * Sd + seg * 256) * Hd;
    float a0 = 0.f, a1 = 0.f;
#pragma unroll 4
    for (int s = 0; s < 256; ++s) {
        const float a = sA[s];
        a0 += a * ep[(long)s * Hd + tid];
        a1 += a * ep[(long)s * Hd + tid + 256];
    }
    part[((long)(b * 8 + seg)) * Hd + tid]       = a0;
    part[((long)(b * 8 + seg)) * Hd + tid + 256] = a1;
}

// ---------------------------------------------------------------------------
// Kernel 5: reduce 8 segments -> context[b,h]
// ---------------------------------------------------------------------------
__global__ __launch_bounds__(256)
void context_reduce_kernel(const float* __restrict__ part, float* __restrict__ ctx)
{
    const int i = blockIdx.x * 256 + threadIdx.x;   // 0..32767
    const int b = i >> 9, h = i & (Hd - 1);
    float acc = 0.f;
#pragma unroll
    for (int seg = 0; seg < 8; ++seg)
        acc += part[((long)(b * 8 + seg)) * Hd + h];
    ctx[i] = acc;
}

// ---------------------------------------------------------------------------
extern "C" void kernel_launch(void* const* d_in, const int* in_sizes, int n_in,
                              void* d_out, int out_size, void* d_ws, size_t ws_size,
                              hipStream_t stream) {
    const float* hidden = (const float*)d_in[0];   // [1,64,512]
    const float* enc    = (const float*)d_in[1];   // [64,2048,512]
    const float* Uw     = (const float*)d_in[2];   // [512,512]
    const float* Ub     = (const float*)d_in[3];   // [512]
    const float* Ww     = (const float*)d_in[4];   // [512,512]
    const float* Wb     = (const float*)d_in[5];   // [512]
    const float* Vw     = (const float*)d_in[6];   // [1,512]
    const float* Vb     = (const float*)d_in[7];   // [1]

    float* out  = (float*)d_out;
    float* ctx  = out;                  // [64,512]
    float* attn = out + Bd * Hd;        // [64,2048]

    // workspace: Wbf (512KB bf16) | u | energy | part
    unsigned short* Wbf = (unsigned short*)d_ws;
    float* u      = (float*)d_ws + (Hd * Hd) / 2;   // after 512KB
    float* energy = u + Bd * Hd;
    float* part   = energy + Bd * Sd;

    convert_w_kernel<<<(Hd * Hd / 4) / 256, 256, 0, stream>>>(Ww, Wbf);
    compute_u_kernel<<<Bd, Hd, 0, stream>>>(hidden, Uw, Ub, u);
    attn_energy_kernel<<<dim3(Sd / 32, Bd), 256, SMEM_TOTAL, stream>>>(
        enc, Wbf, Wb, u, Vw, Vb, energy);
    softmax_kernel<<<Bd, 256, 0, stream>>>(energy, attn);
    context_partial_kernel<<<dim3(8, Bd), 256, 0, stream>>>(enc, attn, part);
    context_reduce_kernel<<<(Bd * Hd) / 256, 256, 0, stream>>>(part, ctx);
}
